// MultiHeadedAttention_34626026341150
// MI455X (gfx1250) — matrix operand order repaired and checked by reading, not compiled, not run on hardware
//
#include <hip/hip_runtime.h>
#include <hip/hip_bf16.h>
#include <stdint.h>

#define BB 8
#define SS 2048
#define DD 768
#define HH 12
#define HDIM 64

typedef __attribute__((ext_vector_type(16))) __bf16 v16bf;
typedef __attribute__((ext_vector_type(8)))  __bf16 v8bf;
typedef __attribute__((ext_vector_type(8)))  float  v8f;

// ---- WMMA helpers ------------------------------------------------------
// A-fragment (16x32 bf16, M x K): lane (l<16) holds M=l%16, K = {8*sel+j} U {16+8*sel+j}
// => two contiguous 16B chunks at (base + 8*sel) and (base + 8*sel + 16), row-major source.
static __device__ __forceinline__ v16bf load_a_frag(const __bf16* p) {
  v8bf lo = *(const v8bf*)p;
  v8bf hi = *(const v8bf*)(p + 16);
  v16bf a;
#pragma unroll
  for (int i = 0; i < 8; ++i) { a[i] = lo[i]; a[i + 8] = hi[i]; }
  return a;
}

// B-fragment (32x16 bf16, K x N): lane holds column N=l%16, K = {8*sel+j} U {16+8*sel+j}
// => two contiguous 16B chunks at (base + 8*sel) and (base + 8*sel + 16), N-major source.
static __device__ __forceinline__ v16bf load_b_frag(const __bf16* p) {
  v8bf lo = *(const v8bf*)p;
  v8bf hi = *(const v8bf*)(p + 16);
  v16bf b;
#pragma unroll
  for (int i = 0; i < 8; ++i) { b[i] = lo[i]; b[i + 8] = hi[i]; }
  return b;
}

static __device__ __forceinline__ v8f wmma_bf16(v16bf a, v16bf b, v8f c) {
  return __builtin_amdgcn_wmma_f32_16x16x32_bf16(false, a, false, b, (short)0, c,
                                                 false, false);
}

// ---- CDNA5 async global->LDS DMA (ASYNCcnt-tracked), per ISA 15.18.3 ---
// Each lane copies 16B: LDS[lds_off(lane)] = MEM[gaddr(lane)].
static __device__ __forceinline__ void async_load_b128(uint32_t lds_off,
                                                       const __bf16* gptr) {
  asm volatile("global_load_async_to_lds_b128 %0, %1, off"
               :
               : "v"(lds_off), "v"(gptr)
               : "memory");
}

static __device__ __forceinline__ void wait_asynccnt0() {
  asm volatile("s_wait_asynccnt 0x0" ::: "memory");
}

// Generic shared pointer -> wave-relative LDS byte offset (aperture
// truncation: keep the low 32 bits, ISA 10.2).
static __device__ __forceinline__ uint32_t lds_off_of(const void* p) {
  return (uint32_t)(uintptr_t)p;
}

// ---- Elementwise cast / transpose kernels ------------------------------
__global__ void k_cast(const float* __restrict__ in, __bf16* __restrict__ out, int n) {
  int i = blockIdx.x * blockDim.x + threadIdx.x;
  if (i < n) out[i] = (__bf16)in[i];
}

// W [H, D, HD] fp32 -> Wt [H, HD, D] bf16 (columns of W become contiguous)
__global__ void k_transpose_head(const float* __restrict__ w, __bf16* __restrict__ wt) {
  int i = blockIdx.x * blockDim.x + threadIdx.x;  // over H*D*HD
  int h = i / (DD * HDIM);
  int rem = i % (DD * HDIM);
  int d = rem / HDIM;
  int e = rem % HDIM;
  wt[((size_t)h * HDIM + e) * DD + d] = (__bf16)w[i];
}

// Wp [D, D] fp32 -> WpT [D, D] bf16
__global__ void k_transpose_wp(const float* __restrict__ w, __bf16* __restrict__ wt) {
  int i = blockIdx.x * blockDim.x + threadIdx.x;  // over D*D
  int r = i / DD, c = i % DD;
  wt[(size_t)c * DD + r] = (__bf16)w[i];
}

// ---- QKV projection: per (b,h): [S,D] x [D,HD] x3 ----------------------
// Block = 8 waves / 128 rows. Weight k-tiles (shared by all 8 waves) are
// staged in LDS with double-buffered async DMA; x A-fragments load direct.
__global__ __launch_bounds__(256) void k_qkv(
    const __bf16* __restrict__ xb,
    const __bf16* __restrict__ wqt, const __bf16* __restrict__ wkt,
    const __bf16* __restrict__ wvt,
    const float* __restrict__ bq, const float* __restrict__ bk,
    const float* __restrict__ bv,
    __bf16* __restrict__ Q, __bf16* __restrict__ K, __bf16* __restrict__ Vt) {
  __shared__ __bf16 sW[2][3][HDIM * 32];  // [buf][q/k/v][n(64) x k(32)] 12KB/buf

  const int tid = threadIdx.x;
  const int bh = blockIdx.x;
  const int b = bh / HH, h = bh % HH;
  const int wave = tid >> 5;
  const int lane = tid & 31;
  const int ml = lane & 15, sel = lane >> 4;
  const int row = blockIdx.y * 128 + wave * 16;

  const __bf16* xrow = xb + ((size_t)b * SS + (row + ml)) * DD;
  const __bf16* wmat[3] = {wqt + (size_t)h * HDIM * DD,
                           wkt + (size_t)h * HDIM * DD,
                           wvt + (size_t)h * HDIM * DD};

  const uint32_t sW_off = lds_off_of(&sW[0][0][0]);
  const int wr = tid >> 2;        // 0..63: weight-tile row (N dim)
  const int wo8 = (tid & 3) * 8;  // element offset inside 32-wide k slice

  // Stage [64 x 32] bf16 tile of each of Wq/Wk/Wv at k0 into buffer `buf`.
  // 256 threads x 3 x 16B = 12KB.
  auto issue_w = [&](int k0, int buf) {
#pragma unroll
    for (int m = 0; m < 3; ++m) {
      async_load_b128(
          sW_off + (uint32_t)(((buf * 3 + m) * HDIM * 32) + wr * 32 + wo8) * 2,
          wmat[m] + (size_t)wr * DD + k0 + wo8);
    }
  };

  const v8f zero = {0, 0, 0, 0, 0, 0, 0, 0};
  v8f aq[4], ak[4], av[4];
#pragma unroll
  for (int n = 0; n < 4; ++n) { aq[n] = zero; ak[n] = zero; av[n] = zero; }

  issue_w(0, 0);
  wait_asynccnt0();
  __syncthreads();

  for (int k0 = 0; k0 < DD; k0 += 32) {
    const int buf = (k0 >> 5) & 1;
    if (k0 + 32 < DD) issue_w(k0 + 32, buf ^ 1);  // DMA next tile behind WMMAs

    v16bf a = load_a_frag(xrow + k0 + 8 * sel);
    const __bf16* tq = sW[buf][0];
    const __bf16* tk = sW[buf][1];
    const __bf16* tv = sW[buf][2];
#pragma unroll
    for (int n = 0; n < 4; ++n) {
      const int fo = (n * 16 + ml) * 32 + 8 * sel;
      aq[n] = wmma_bf16(a, load_b_frag(tq + fo), aq[n]);
      ak[n] = wmma_bf16(a, load_b_frag(tk + fo), ak[n]);
      av[n] = wmma_bf16(a, load_b_frag(tv + fo), av[n]);
    }

    wait_asynccnt0();
    __syncthreads();
  }

  // C-layout: lane holds col = n*16 + ml, rows r + 8*sel (r = 0..7)
  __bf16* qout = Q + (size_t)bh * SS * HDIM;   // [S, HD] row-major
  __bf16* kout = K + (size_t)bh * SS * HDIM;   // [S, HD] row-major
  __bf16* vout = Vt + (size_t)bh * HDIM * SS;  // [HD, S]  (transposed!)
#pragma unroll
  for (int n = 0; n < 4; ++n) {
    const int e = n * 16 + ml;
    const float bqe = bq[h * HDIM + e];
    const float bke = bk[h * HDIM + e];
    const float bve = bv[h * HDIM + e];
#pragma unroll
    for (int r = 0; r < 8; ++r) {
      const int s = row + r + 8 * sel;
      qout[(size_t)s * HDIM + e] = (__bf16)(aq[n][r] + bqe);
      kout[(size_t)s * HDIM + e] = (__bf16)(ak[n][r] + bke);
      vout[(size_t)e * SS + s]   = (__bf16)(av[n][r] + bve);
    }
  }
}

// ---- Flash attention: per (b,h), 64 query rows per block (4 waves) -----
// K/V tiles staged in LDS via async DMA (double-buffered): all 4 waves share
// the same 32-key tile, so staging cuts VMEM traffic 4x.
__global__ __launch_bounds__(128) void k_attn(
    const __bf16* __restrict__ Q, const __bf16* __restrict__ K,
    const __bf16* __restrict__ Vt, __bf16* __restrict__ CTX) {
  __shared__ __bf16 sK[2][32 * HDIM];   // [key_local][hd]   4KB per buffer
  __shared__ __bf16 sV[2][HDIM * 32];   // [hd][key_local]   4KB per buffer
  __shared__ __bf16 lds_p[4][16 * 32];  // wave-private P tile

  const int tid = threadIdx.x;
  const int bh = blockIdx.x;
  const int b = bh / HH, h = bh % HH;
  const int wave = tid >> 5;
  const int lane = tid & 31;
  const int ml = lane & 15, sel = lane >> 4;
  const int q0 = blockIdx.y * 64 + wave * 16;

  const __bf16* qbase = Q + (size_t)bh * SS * HDIM;
  const __bf16* kbase = K + (size_t)bh * SS * HDIM;
  const __bf16* vbase = Vt + (size_t)bh * HDIM * SS;

  const uint32_t sK_off = lds_off_of(&sK[0][0]);
  const uint32_t sV_off = lds_off_of(&sV[0][0]);

  // Cooperative async stage of one 32-key K tile + V tile into buffer `buf`.
  // 128 threads x 2 chunks x 16B = 4KB each for K and V.
  auto issue_tile = [&](int kt, int buf) {
#pragma unroll
    for (int j = 0; j < 2; ++j) {
      const int c = tid + 128 * j;  // chunk id 0..255
      const int kr = c >> 3, ko = (c & 7) * 8;
      async_load_b128(sK_off + (uint32_t)(buf * 32 * HDIM + kr * HDIM + ko) * 2,
                      kbase + (size_t)(kt + kr) * HDIM + ko);
      const int vr = c >> 2, vo = (c & 3) * 8;
      async_load_b128(sV_off + (uint32_t)(buf * HDIM * 32 + vr * 32 + vo) * 2,
                      vbase + (size_t)vr * SS + kt + vo);
    }
  };

  // Q A-fragments, kept for the whole key loop (HD=64 -> 2 k-steps)
  const v16bf aq0 = load_a_frag(qbase + (size_t)(q0 + ml) * HDIM + 0 + 8 * sel);
  const v16bf aq1 = load_a_frag(qbase + (size_t)(q0 + ml) * HDIM + 32 + 8 * sel);

  // softmax scale folded with log2(e) so we can use exp2
  const float sc = 1.44269504088896f / (8.0f + 1e-6f);

  float mrow[8], lrow[8];
  const v8f zero = {0, 0, 0, 0, 0, 0, 0, 0};
  v8f acc[4];
#pragma unroll
  for (int r = 0; r < 8; ++r) { mrow[r] = -1e30f; lrow[r] = 0.0f; }
#pragma unroll
  for (int n = 0; n < 4; ++n) acc[n] = zero;

  __bf16* lp = lds_p[wave];

  issue_tile(0, 0);
  wait_asynccnt0();
  __syncthreads();

  for (int kt = 0; kt < SS; kt += 32) {
    const int buf = (kt >> 5) & 1;
    // kick off DMA for the next tile into the other buffer (wraps harmlessly)
    issue_tile((kt + 32) & (SS - 1), buf ^ 1);

    const __bf16* tK = sK[buf];
    const __bf16* tV = sV[buf];

    // S = Q * K^T for 32 keys (two 16-col n-tiles); contraction over HD
    v8f s0 = zero, s1 = zero;
    {
      const __bf16* pk0 = tK + (size_t)ml * HDIM + 8 * sel;
      s0 = wmma_bf16(aq0, load_b_frag(pk0), s0);
      s0 = wmma_bf16(aq1, load_b_frag(pk0 + 32), s0);
      const __bf16* pk1 = pk0 + (size_t)16 * HDIM;
      s1 = wmma_bf16(aq0, load_b_frag(pk1), s1);
      s1 = wmma_bf16(aq1, load_b_frag(pk1 + 32), s1);
    }

    // online softmax (per row; rows live in VGPR index r, cols across 16 lanes)
#pragma unroll
    for (int r = 0; r < 8; ++r) {
      float a = s0[r] * sc;
      float c = s1[r] * sc;
      float t = fmaxf(a, c);
      t = fmaxf(t, __shfl_xor(t, 1));
      t = fmaxf(t, __shfl_xor(t, 2));
      t = fmaxf(t, __shfl_xor(t, 4));
      t = fmaxf(t, __shfl_xor(t, 8));
      const float nm = fmaxf(mrow[r], t);
      const float corr = exp2f(mrow[r] - nm);
      mrow[r] = nm;
      const float p0 = exp2f(a - nm);
      const float p1 = exp2f(c - nm);
      s0[r] = p0;
      s1[r] = p1;
      float rs = p0 + p1;
      rs += __shfl_xor(rs, 1);
      rs += __shfl_xor(rs, 2);
      rs += __shfl_xor(rs, 4);
      rs += __shfl_xor(rs, 8);
      lrow[r] = lrow[r] * corr + rs;
      acc[0][r] *= corr;
      acc[1][r] *= corr;
      acc[2][r] *= corr;
      acc[3][r] *= corr;
    }

    // C-layout P -> row-major 16x32 bf16 in wave-private LDS
#pragma unroll
    for (int r = 0; r < 8; ++r) {
      const int rr = r + 8 * sel;
      lp[rr * 32 + ml] = (__bf16)s0[r];
      lp[rr * 32 + 16 + ml] = (__bf16)s1[r];
    }
    __builtin_amdgcn_sched_barrier(0);  // DS ops are in-order within a wave
    const v16bf ap = load_a_frag(lp + ml * 32 + 8 * sel);

    // ctx += P * V ; contraction over 32 keys, V tile is [hd][key] in LDS
#pragma unroll
    for (int n = 0; n < 4; ++n) {
      const __bf16* pv = tV + (size_t)(n * 16 + ml) * 32 + 8 * sel;
      acc[n] = wmma_bf16(ap, load_b_frag(pv), acc[n]);
    }

    // next tile's DMA must be complete and all waves done before reuse
    wait_asynccnt0();
    __syncthreads();
  }

  // normalize and store ctx (bf16) at [b, s, h*64 + e]
  __bf16* cbase = CTX + (size_t)b * SS * DD + (size_t)h * HDIM;
#pragma unroll
  for (int n = 0; n < 4; ++n) {
#pragma unroll
    for (int r = 0; r < 8; ++r) {
      const float v = acc[n][r] / lrow[r];
      cbase[(size_t)(q0 + r + 8 * sel) * DD + n * 16 + ml] = (__bf16)v;
    }
  }
}

// ---- Output projection: [B*S, D] x [D, D] + bp, fp32 out ---------------
// Weight k-tiles staged via double-buffered async DMA (shared by 8 waves).
__global__ __launch_bounds__(256) void k_proj(
    const __bf16* __restrict__ CTX, const __bf16* __restrict__ WPT,
    const float* __restrict__ bp, float* __restrict__ out) {
  __shared__ __bf16 sW[2][HDIM * 32];  // [buf][n(64) x k(32)] 4KB per buffer

  const int tid = threadIdx.x;
  const int wave = tid >> 5, lane = tid & 31;
  const int ml = lane & 15, sel = lane >> 4;
  const int row = blockIdx.x * 128 + wave * 16;
  const int col0 = blockIdx.y * 64;

  const __bf16* crow = CTX + (size_t)(row + ml) * DD;
  const __bf16* wp = WPT + (size_t)col0 * DD;  // rows col0..col0+63 of WpT

  const uint32_t sW_off = lds_off_of(&sW[0][0]);
  const int wr = tid >> 2;        // 0..63
  const int wo8 = (tid & 3) * 8;  // element offset

  auto issue_w = [&](int k0, int buf) {
    async_load_b128(sW_off + (uint32_t)(buf * HDIM * 32 + wr * 32 + wo8) * 2,
                    wp + (size_t)wr * DD + k0 + wo8);
  };

  const v8f zero = {0, 0, 0, 0, 0, 0, 0, 0};
  v8f acc[4];
#pragma unroll
  for (int n = 0; n < 4; ++n) acc[n] = zero;

  issue_w(0, 0);
  wait_asynccnt0();
  __syncthreads();

  for (int k0 = 0; k0 < DD; k0 += 32) {
    const int buf = (k0 >> 5) & 1;
    if (k0 + 32 < DD) issue_w(k0 + 32, buf ^ 1);

    v16bf a = load_a_frag(crow + k0 + 8 * sel);
    const __bf16* tw = sW[buf];
#pragma unroll
    for (int n = 0; n < 4; ++n) {
      acc[n] = wmma_bf16(a, load_b_frag(tw + (n * 16 + ml) * 32 + 8 * sel), acc[n]);
    }

    wait_asynccnt0();
    __syncthreads();
  }

#pragma unroll
  for (int n = 0; n < 4; ++n) {
    const int c = col0 + n * 16 + ml;
    const float bpc = bp[c];
#pragma unroll
    for (int r = 0; r < 8; ++r) {
      out[(size_t)(row + r + 8 * sel) * DD + c] = acc[n][r] + bpc;
    }
  }
}

// ---- Launch ------------------------------------------------------------
extern "C" void kernel_launch(void* const* d_in, const int* in_sizes, int n_in,
                              void* d_out, int out_size, void* d_ws, size_t ws_size,
                              hipStream_t stream) {
  (void)in_sizes; (void)n_in; (void)out_size; (void)ws_size;

  const float* x  = (const float*)d_in[0];
  const float* Wq = (const float*)d_in[1];
  const float* bq = (const float*)d_in[2];
  const float* Wk = (const float*)d_in[3];
  const float* bk = (const float*)d_in[4];
  const float* Wv = (const float*)d_in[5];
  const float* bv = (const float*)d_in[6];
  const float* Wp = (const float*)d_in[7];
  const float* bp = (const float*)d_in[8];
  float* out = (float*)d_out;

  const size_t NXB = (size_t)BB * SS * DD;   // 12,582,912 (also = B*H*S*HD)
  const size_t NWH = (size_t)HH * DD * HDIM; // 589,824
  const size_t NWP = (size_t)DD * DD;        // 589,824

  __bf16* wsb = (__bf16*)d_ws;
  __bf16* XB  = wsb;
  __bf16* WQT = XB + NXB;
  __bf16* WKT = WQT + NWH;
  __bf16* WVT = WKT + NWH;
  __bf16* WPT = WVT + NWH;
  __bf16* Qb  = WPT + NWP;
  __bf16* Kb  = Qb + NXB;
  __bf16* Vtb = Kb + NXB;
  __bf16* CTX = Vtb + NXB;
  // total ws usage ~125 MB (bf16)

  k_cast<<<(int)(NXB / 256), 256, 0, stream>>>(x, XB, (int)NXB);
  k_transpose_head<<<(int)(NWH / 256), 256, 0, stream>>>(Wq, WQT);
  k_transpose_head<<<(int)(NWH / 256), 256, 0, stream>>>(Wk, WKT);
  k_transpose_head<<<(int)(NWH / 256), 256, 0, stream>>>(Wv, WVT);
  k_transpose_wp<<<(int)(NWP / 256), 256, 0, stream>>>(Wp, WPT);

  k_qkv<<<dim3(BB * HH, SS / 128), 256, 0, stream>>>(XB, WQT, WKT, WVT, bq, bk, bv,
                                                     Qb, Kb, Vtb);
  k_attn<<<dim3(BB * HH, SS / 64), 128, 0, stream>>>(Qb, Kb, Vtb, CTX);
  k_proj<<<dim3((BB * SS) / 128, DD / 64), 256, 0, stream>>>(CTX, WPT, bp, out);
}
